// Attention_73358041416045
// MI455X (gfx1250) — matrix operand order repaired and checked by reading, compiled, not run on hardware
//
#include <hip/hip_runtime.h>

typedef __bf16 bf16;
typedef __bf16 bf16x4  __attribute__((ext_vector_type(4)));
typedef __bf16 bf16x8  __attribute__((ext_vector_type(8)));
typedef __bf16 bf16x16 __attribute__((ext_vector_type(16)));
typedef float  f32x8   __attribute__((ext_vector_type(8)));
typedef unsigned int u32x4 __attribute__((ext_vector_type(4)));
typedef int i32x4 __attribute__((ext_vector_type(4)));
typedef int i32x8 __attribute__((ext_vector_type(8)));

constexpr int Bc = 2, Sc = 2048, Dc = 2048, Hc = 16, HDc = 128;
constexpr float SCALE = 0.08838834764831845f;  // 1/sqrt(128)

// LDS row stride for staged 32-element bf16 rows: 64B data + 16B TDM pad = 80B
// -> 20-dword stride, 16 fragment lanes map to distinct banks; rows stay 16B aligned.
constexpr int LDS_ROW_B = 80;
constexpr int LDS_TILE_B = 128 * LDS_ROW_B;   // one 128x32 bf16 tile: 10240 B
constexpr int LDS_BUF_B  = 2 * LDS_TILE_B;    // A tile + B tile per buffer

#define WMMA_BF16(a, b, c) \
  __builtin_amdgcn_wmma_f32_16x16x32_bf16(false, (a), false, (b), (short)0, (c), false, false)

__device__ __forceinline__ bf16x8 ld8(const bf16* p) {
  return *reinterpret_cast<const bf16x8*>(p);
}
__device__ __forceinline__ bf16x16 cat16(bf16x8 lo, bf16x8 hi) {
  return __builtin_shufflevector(lo, hi, 0,1,2,3,4,5,6,7,8,9,10,11,12,13,14,15);
}
// A-frag (16x32 bf16): per-lane chunks at elements [kbase, kbase+8) and [kbase+16, kbase+24)
__device__ __forceinline__ bf16x16 ldA(const bf16* p) { return cat16(ld8(p), ld8(p + 16)); }
// B-frag (32x16 bf16): per-lane chunks at elements [kbase, kbase+8) and [kbase+16, kbase+24)
__device__ __forceinline__ bf16x16 ldB(const bf16* p) { return cat16(ld8(p), ld8(p + 16)); }

__device__ __forceinline__ float rmax16(float v) {
  v = fmaxf(v, __shfl_xor(v, 1));
  v = fmaxf(v, __shfl_xor(v, 2));
  v = fmaxf(v, __shfl_xor(v, 4));
  v = fmaxf(v, __shfl_xor(v, 8));
  return v;
}
__device__ __forceinline__ float rsum16(float v) {
  v += __shfl_xor(v, 1);
  v += __shfl_xor(v, 2);
  v += __shfl_xor(v, 4);
  v += __shfl_xor(v, 8);
  return v;
}

// ---------------------------------------------------------------- TDM tile load
// 2-D tile load: 128 rows x 32 bf16 columns from a row-major [rows, K] bf16 matrix
// into LDS at lds_addr, padded 16B per 64B row (LDS row stride 80B).
__device__ __forceinline__ void tdm_load_tile(unsigned lds_addr, const bf16* gptr,
                                              int tensor_rows, int K) {
  unsigned long long ga = (unsigned long long)(size_t)gptr;
  u32x4 g0;
  g0[0] = 1u;                                   // count=1, user descriptor
  g0[1] = lds_addr;                             // LDS byte address
  g0[2] = (unsigned)(ga & 0xffffffffull);       // global_addr[31:0]
  g0[3] = (unsigned)((ga >> 32) & 0x01ffffffull) | (2u << 30);  // addr[56:32] | type=2
  i32x8 g1;
  g1[0] = (int)((1u << 16)      // data_size = 2 bytes
              | (1u << 20)      // pad_enable
              | (3u << 22)      // pad_interval: 16 dwords (64B)
              | (3u << 25));    // pad_amount: 4 dwords (16B)
  g1[1] = (int)(((unsigned)K & 0xffffu) << 16);                 // tensor_dim0[15:0]
  g1[2] = (int)((((unsigned)K >> 16) & 0xffffu) |
                (((unsigned)tensor_rows & 0xffffu) << 16));     // dim0[31:16] | dim1[15:0]
  g1[3] = (int)(((((unsigned)tensor_rows) >> 16) & 0xffffu) |
                (32u << 16));                                   // dim1[31:16] | tile_dim0=32
  g1[4] = 128;                                                  // tile_dim1=128, tile_dim2=0
  g1[5] = K;                                                    // tensor_dim0_stride[31:0]
  g1[6] = 0;                                                    // stride0[47:32] | stride1[15:0]
  g1[7] = 0;                                                    // stride1[47:16]
  i32x4 z4 = {0, 0, 0, 0};
#if defined(__clang_major__) && (__clang_major__ >= 23)
  i32x8 z8 = {0, 0, 0, 0, 0, 0, 0, 0};
  __builtin_amdgcn_tensor_load_to_lds(g0, g1, z4, z4, z8, 0);
#else
  __builtin_amdgcn_tensor_load_to_lds(g0, g1, z4, z4, 0);
#endif
}

// ---------------------------------------------------------------- convert
__global__ __launch_bounds__(256) void f32_to_bf16_kernel(const float* __restrict__ in,
                                                          bf16* __restrict__ out, int n) {
  int i = (blockIdx.x * 256 + threadIdx.x) * 4;
  if (i < n) {
    float4 v = *reinterpret_cast<const float4*>(in + i);
    bf16x4 o;
    o[0] = (bf16)v.x; o[1] = (bf16)v.y; o[2] = (bf16)v.z; o[3] = (bf16)v.w;
    *reinterpret_cast<bf16x4*>(out + i) = o;
  }
}

// ---------------------------------------------------------------- GEMM: out = A * W^T
// MODE 0: store bf16 row-major [M,N]
// MODE 1: store bf16 transposed into vT[b,h,hd,s]
// MODE 2: store f32 row-major [M,N]
template <int MODE>
__device__ __forceinline__ void store_tile(const f32x8& c, int mb, int nb, int lane,
                                           bf16* outb, float* outf, int N) {
  const int half = lane >> 4, nloc = lane & 15;
  if constexpr (MODE == 2) {
#pragma unroll
    for (int r = 0; r < 8; ++r) {
      int row = mb + half * 8 + r;
      outf[(size_t)row * N + nb + nloc] = c[r];
    }
  } else if constexpr (MODE == 0) {
#pragma unroll
    for (int r = 0; r < 8; ++r) {
      int row = mb + half * 8 + r;
      outb[(size_t)row * N + nb + nloc] = (bf16)c[r];
    }
  } else {  // MODE == 1 : vT store, per-lane 8 consecutive s values -> one b128
    int col = nb + nloc;
    int b = mb >> 11;                       // S = 2048
    int sbase = (mb & (Sc - 1)) + half * 8;
    int h = col >> 7, hd = col & (HDc - 1); // HD = 128
    bf16x8 pk;
#pragma unroll
    for (int r = 0; r < 8; ++r) pk[r] = (bf16)c[r];
    *reinterpret_cast<bf16x8*>(outb + (((size_t)b * Hc + h) * HDc + hd) * Sc + sbase) = pk;
  }
}

// Workgroup: 128 threads (4 waves). WG tile 128x128, 64x64 per wave (16 WMMAs/K-step).
// Wave 0 double-buffers 128x32 A and B tiles into LDS via the Tensor Data Mover.
template <int MODE>
__global__ __launch_bounds__(128) void wmma_gemm_tdm_kernel(const bf16* __restrict__ A,
                                                            const bf16* __restrict__ W,
                                                            bf16* __restrict__ outb,
                                                            float* __restrict__ outf,
                                                            int Mtot, int N, int K) {
  __shared__ alignas(16) char smem[2 * LDS_BUF_B];  // 40 KB: 2 buffers x (A tile + B tile)

  const int lane = threadIdx.x & 31;
  const int wave = threadIdx.x >> 5;
  const int half = lane >> 4;
  const int nloc = lane & 15;
  const int wr = wave >> 1, wc = wave & 1;

  const int WGm = blockIdx.y * 128;
  const int WGn = blockIdx.x * 128;

  const unsigned lds_base = (unsigned)(size_t)(void*)smem;  // generic low 32 bits = LDS offset
  const int nsteps = K / 32;

  f32x8 acc[4][4];
#pragma unroll
  for (int ti = 0; ti < 4; ++ti)
#pragma unroll
    for (int tj = 0; tj < 4; ++tj) acc[ti][tj] = f32x8{};

  if (wave == 0) {
    tdm_load_tile(lds_base, A + (size_t)WGm * K, Mtot, K);
    tdm_load_tile(lds_base + LDS_TILE_B, W + (size_t)WGn * K, N, K);
  }

  for (int i = 0; i < nsteps; ++i) {
    if (wave == 0) {
      if (i + 1 < nsteps) {
        int k1 = (i + 1) * 32;
        unsigned lb = lds_base + (unsigned)(((i + 1) & 1) * LDS_BUF_B);
        tdm_load_tile(lb, A + (size_t)WGm * K + k1, Mtot, K);
        tdm_load_tile(lb + LDS_TILE_B, W + (size_t)WGn * K + k1, N, K);
        __builtin_amdgcn_s_wait_tensorcnt(2);  // pair for step i complete
      } else {
        __builtin_amdgcn_s_wait_tensorcnt(0);
      }
    }
    __syncthreads();

    const char* aB = smem + (i & 1) * LDS_BUF_B;
    const char* bB = aB + LDS_TILE_B;

    bf16x16 af[4];
#pragma unroll
    for (int ti = 0; ti < 4; ++ti) {
      const bf16* p = (const bf16*)(aB + (wr * 64 + ti * 16 + nloc) * LDS_ROW_B + half * 16);
      af[ti] = cat16(ld8(p), ld8(p + 16));  // chunks at +0B and +32B
    }
#pragma unroll
    for (int tj = 0; tj < 4; ++tj) {
      const bf16* p = (const bf16*)(bB + (wc * 64 + tj * 16 + nloc) * LDS_ROW_B + half * 16);
      bf16x16 bfrag = cat16(ld8(p), ld8(p + 16));  // chunks at +0B and +32B
#pragma unroll
      for (int ti = 0; ti < 4; ++ti) acc[ti][tj] = WMMA_BF16(af[ti], bfrag, acc[ti][tj]);
    }
    __syncthreads();
  }

#pragma unroll
  for (int ti = 0; ti < 4; ++ti)
#pragma unroll
    for (int tj = 0; tj < 4; ++tj)
      store_tile<MODE>(acc[ti][tj], WGm + wr * 64 + ti * 16, WGn + wc * 64 + tj * 16,
                       lane, outb, outf, N);
}

// ---------------------------------------------------------------- flash attention
// One wave per (b,h, 16-query tile). Causal. q,k in [b*s, H*HD] bf16; v transposed [b,h,hd,s].
__global__ __launch_bounds__(32) void flash_attn_kernel(const bf16* __restrict__ q,
                                                        const bf16* __restrict__ k,
                                                        const bf16* __restrict__ vt,
                                                        bf16* __restrict__ ctx) {
  __shared__ alignas(16) bf16 pLds[16 * 32];

  const int lane = threadIdx.x & 31;
  const int half = lane >> 4;
  const int nloc = lane & 15;
  const int qtile = blockIdx.x;
  const int b = blockIdx.y >> 4;
  const int h = blockIdx.y & 15;

  bf16x16 qf[4];
  {
    const bf16* qp = q + ((size_t)b * Sc + qtile * 16 + nloc) * Dc + h * HDc + half * 8;
#pragma unroll
    for (int ks = 0; ks < 4; ++ks) qf[ks] = ldA(qp + ks * 32);
  }

  f32x8 oacc[8];
#pragma unroll
  for (int n = 0; n < 8; ++n) oacc[n] = f32x8{};
  float mrun[8], lrun[8];
#pragma unroll
  for (int r = 0; r < 8; ++r) { mrun[r] = -1e30f; lrun[r] = 0.0f; }

  const int nsteps = ((qtile * 16 + 15) >> 5) + 1;

  for (int kb = 0; kb < nsteps; ++kb) {
    const int key0 = kb * 32;

    f32x8 s0 = {}, s1 = {};
    {
      const bf16* kp0 = k + ((size_t)b * Sc + key0 + nloc) * Dc + h * HDc + half * 8;
      const bf16* kp1 = kp0 + (size_t)16 * Dc;
#pragma unroll
      for (int ks = 0; ks < 4; ++ks) {
        bf16x16 bk0 = ldB(kp0 + ks * 32);
        bf16x16 bk1 = ldB(kp1 + ks * 32);
        s0 = WMMA_BF16(qf[ks], bk0, s0);
        s1 = WMMA_BF16(qf[ks], bk1, s1);
      }
    }

    const int keyc0 = key0 + nloc;
    const int keyc1 = keyc0 + 16;
#pragma unroll
    for (int r = 0; r < 8; ++r) {
      const int qrow = qtile * 16 + half * 8 + r;
      float v0 = s0[r] * SCALE + (keyc0 > qrow ? -1e9f : 0.0f);
      float v1 = s1[r] * SCALE + (keyc1 > qrow ? -1e9f : 0.0f);
      float rowmax = rmax16(fmaxf(v0, v1));
      float newm = fmaxf(mrun[r], rowmax);
      float p0 = __expf(v0 - newm);
      float p1 = __expf(v1 - newm);
      float alpha = __expf(mrun[r] - newm);
      float rowsum = rsum16(p0 + p1);
      lrun[r] = lrun[r] * alpha + rowsum;
      mrun[r] = newm;
#pragma unroll
      for (int n = 0; n < 8; ++n) oacc[n][r] *= alpha;
      pLds[(half * 8 + r) * 32 + nloc] = (bf16)p0;
      pLds[(half * 8 + r) * 32 + 16 + nloc] = (bf16)p1;
    }
    __syncthreads();

    bf16x16 pf = ldA(&pLds[nloc * 32 + half * 8]);

    const bf16* vp = vt + (((size_t)b * Hc + h) * HDc + nloc) * Sc + key0 + half * 8;
#pragma unroll
    for (int n = 0; n < 8; ++n) {
      bf16x16 bv = ldB(vp + (size_t)n * 16 * Sc);
      oacc[n] = WMMA_BF16(pf, bv, oacc[n]);
    }
    __syncthreads();
  }

  float inv[8];
#pragma unroll
  for (int r = 0; r < 8; ++r) inv[r] = 1.0f / lrun[r];
  bf16* cp = ctx + ((size_t)b * Sc + qtile * 16) * Dc + h * HDc;
#pragma unroll
  for (int n = 0; n < 8; ++n) {
#pragma unroll
    for (int r = 0; r < 8; ++r) {
      cp[(size_t)(half * 8 + r) * Dc + n * 16 + nloc] = (bf16)(oacc[n][r] * inv[r]);
    }
  }
}

// ---------------------------------------------------------------- launch
extern "C" void kernel_launch(void* const* d_in, const int* in_sizes, int n_in,
                              void* d_out, int out_size, void* d_ws, size_t ws_size,
                              hipStream_t stream) {
  (void)in_sizes; (void)n_in; (void)out_size; (void)ws_size;
  const float* x  = (const float*)d_in[0];
  // d_in[1] = freqs_complex (unused: rotary emb is a no-op in the reference)
  // d_in[2] = mask (causal, reproduced analytically)
  const float* wq = (const float*)d_in[3];
  const float* wk = (const float*)d_in[4];
  const float* wv = (const float*)d_in[5];
  const float* wo = (const float*)d_in[6];
  float* out = (float*)d_out;

  const size_t MT = (size_t)Bc * Sc;  // 4096 tokens
  char* ws = (char*)d_ws;
  size_t off = 0;
  auto alloc = [&](size_t elems) -> bf16* {
    bf16* p = (bf16*)(ws + off);
    off += elems * sizeof(bf16);
    return p;
  };
  bf16* xb   = alloc(MT * Dc);
  bf16* qb   = alloc(MT * Dc);
  bf16* kbuf = alloc(MT * Dc);
  bf16* vtb  = alloc(MT * Dc);   // [B, H, HD, S]
  bf16* ctx  = alloc(MT * Dc);
  bf16* wqb  = alloc((size_t)Dc * Dc);
  bf16* wkb  = alloc((size_t)Dc * Dc);
  bf16* wvb  = alloc((size_t)Dc * Dc);
  bf16* wob  = alloc((size_t)Dc * Dc);

  const int nx = (int)(MT * Dc);      // 8,388,608
  const int nw = Dc * Dc;             // 4,194,304
  f32_to_bf16_kernel<<<nx / 1024, 256, 0, stream>>>(x,  xb,  nx);
  f32_to_bf16_kernel<<<nw / 1024, 256, 0, stream>>>(wq, wqb, nw);
  f32_to_bf16_kernel<<<nw / 1024, 256, 0, stream>>>(wk, wkb, nw);
  f32_to_bf16_kernel<<<nw / 1024, 256, 0, stream>>>(wv, wvb, nw);
  f32_to_bf16_kernel<<<nw / 1024, 256, 0, stream>>>(wo, wob, nw);

  dim3 gemm_grid(Dc / 128, (int)(MT / 128));  // (16, 32)
  wmma_gemm_tdm_kernel<0><<<gemm_grid, 128, 0, stream>>>(xb, wqb, qb,   nullptr, (int)MT, Dc, Dc);
  wmma_gemm_tdm_kernel<0><<<gemm_grid, 128, 0, stream>>>(xb, wkb, kbuf, nullptr, (int)MT, Dc, Dc);
  wmma_gemm_tdm_kernel<1><<<gemm_grid, 128, 0, stream>>>(xb, wvb, vtb,  nullptr, (int)MT, Dc, Dc);

  flash_attn_kernel<<<dim3(Sc / 16, Bc * Hc), 32, 0, stream>>>(qb, kbuf, vtb, ctx);

  wmma_gemm_tdm_kernel<2><<<gemm_grid, 128, 0, stream>>>(ctx, wob, nullptr, out, (int)MT, Dc, Dc);
}
